// RetinaNet_34583076667485
// MI455X (gfx1250) — matrix operand order repaired and checked by reading, not compiled, not run on hardware
//
#include <hip/hip_runtime.h>
#include <hip/hip_bf16.h>

typedef __attribute__((ext_vector_type(16))) _Float16 v16h;
typedef __attribute__((ext_vector_type(8)))  _Float16 v8h;
typedef __attribute__((ext_vector_type(8)))  float    v8f;

typedef __attribute__((address_space(3))) _Float16 lds_h;

// ---------------------------------------------------------------------------
// Async global -> LDS 128-bit copy (CDNA5 direct path, ASYNCcnt-tracked).
// Inline asm per CDNA5 ISA (portable across toolchains):
//   VGLOBAL encoding: vdst = 32-bit LDS byte offset, vaddr = 64-bit address.
// ---------------------------------------------------------------------------
__device__ __forceinline__ void async_g2l_b128(const _Float16* g, _Float16* l) {
    unsigned loff = (unsigned)(unsigned long long)(lds_h*)l;   // AS3 ptr == LDS offset
    asm volatile("global_load_async_to_lds_b128 %0, %1, off"
                 :: "v"(loff), "v"(g) : "memory");
}

__device__ __forceinline__ void wait_asynccnt0() {
#if __has_builtin(__builtin_amdgcn_s_wait_asynccnt)
    __builtin_amdgcn_s_wait_asynccnt(0);
#else
    asm volatile("s_wait_asynccnt 0" ::: "memory");
#endif
}

// ---------------------------------------------------------------------------
// Zero-fill (for padded-halo f16 buffers). n8 = count of 8-f16 chunks.
// ---------------------------------------------------------------------------
__global__ void fill0_f16(_Float16* __restrict__ p, size_t n8) {
    size_t i = (size_t)blockIdx.x * blockDim.x + threadIdx.x;
    if (i < n8) {
        v8h z = {};
        ((v8h*)p)[i] = z;
    }
}

// ---------------------------------------------------------------------------
// Weight conversion: fp32 [Co][128][3][3][3]  ->  f16 B-fragment order
//   out[(((t*nCoT + cot)*4 + cib)*32 + lane)*16 + e]
//   co = cot*16 + (lane&15);  ci = cib*32 + e + (lane&16 ? 16 : 0)
// B operand (32x16 KxN) in contiguous k order k = 16*(lane/16) + e, same as A, so the
// conv kernel loads one contiguous 32B chunk per lane per fragment.
// ---------------------------------------------------------------------------
__global__ void wconv_frag(const float* __restrict__ w, _Float16* __restrict__ out,
                           int Co, int nCoT) {
    int idx = blockIdx.x * blockDim.x + threadIdx.x;
    int total = 27 * nCoT * 2048;
    if (idx >= total) return;
    int e    = idx & 15;
    int lane = (idx >> 4) & 31;
    int cib  = (idx >> 9) & 3;
    int cot  = (idx >> 11) % nCoT;
    int t    = idx / (nCoT << 11);
    int co   = cot * 16 + (lane & 15);
    int ci   = cib * 32 + e + ((lane & 16) ? 16 : 0);
    float v = 0.f;
    if (co < Co) v = w[((size_t)co * 128 + ci) * 27 + t];
    out[idx] = (_Float16)v;
}

// ---------------------------------------------------------------------------
// Feature conversion: fp32 NCDHW -> f16 padded channels-last
// ---------------------------------------------------------------------------
__global__ void im2half(const float* __restrict__ in, _Float16* __restrict__ out, int Dd) {
    size_t idx = (size_t)blockIdx.x * blockDim.x + threadIdx.x;
    size_t total = (size_t)2 * 128 * Dd * Dd * Dd;
    if (idx >= total) return;
    int c   = (int)(idx & 127);
    size_t vox = idx >> 7;
    int w = (int)(vox % Dd);
    int h = (int)((vox / Dd) % Dd);
    int d = (int)((vox / ((size_t)Dd * Dd)) % Dd);
    int n = (int)(vox / ((size_t)Dd * Dd * Dd));
    float v = in[((((size_t)n * 128 + c) * Dd + d) * Dd + h) * Dd + w];
    int P = Dd + 2;
    size_t pv = (((size_t)(n * P + d + 1) * P + (h + 1)) * P + (w + 1));
    out[pv * 128 + c] = (_Float16)v;
}

// ---------------------------------------------------------------------------
// Implicit-GEMM 3D conv via WMMA f32_16x16x32_f16.
//   One workgroup (8 waves) = one (n,d,h) output row (W voxels) x 128 co.
//   Wave w owns co-tile w (16 channels), MT accumulators of 16 voxels each.
//   K-loop: 27 taps x 4 ci-blocks of 32.  A-tile double-buffered in LDS via
//   async global->LDS copies; B streamed from fragment-ordered weights.
// ---------------------------------------------------------------------------
template <int MT>
__global__ void __launch_bounds__(256)
conv3d_wmma(const _Float16* __restrict__ inPad,
            const _Float16* __restrict__ wfrag,   // [27][nCoT][4][32][16] f16
            const float*    __restrict__ bias,
            _Float16*       __restrict__ outPad,  // tower mode (may be null)
            float*          __restrict__ outF32,  // out-conv mode (may be null)
            int D, int H, int W, int nCoT, int Co, int relu) {
    __shared__ __align__(16) _Float16 Alds[2][MT * 16 * 128];

    const int tid  = threadIdx.x;
    const int wave = tid >> 5;
    const int lane = tid & 31;

    int b = blockIdx.x;
    const int h = b % H;
    const int d = (b / H) % D;
    const int n = b / (H * D);

    const int P  = W + 2;            // padded extent
    const int PH = H + 2;
    const int PD = D + 2;

    // accumulators pre-loaded with bias (same across the 8 rows of each tile)
    v8f acc[MT];
    {
        float bv = (wave < nCoT) ? bias[wave * 16 + (lane & 15)] : 0.f;
#pragma unroll
        for (int mt = 0; mt < MT; ++mt)
#pragma unroll
            for (int i = 0; i < 8; ++i) acc[mt][i] = bv;
    }

    // stage A tile for tap `t` into LDS buffer `bufSel` (async, 16B/chunk)
    auto stageA = [&](int t, int bufSel) {
        const int dz = t / 9, dy = (t / 3) % 3, dx = t % 3;
#pragma unroll
        for (int c = 0; c < MT; ++c) {
            int cc = c * 256 + tid;        // chunk id
            int m  = cc >> 4;              // voxel row within tile
            int ci = (cc & 15) << 3;       // 8-f16 chunk of channels
            int wv = m + dx;               // padded w coordinate
            _Float16* l = &Alds[bufSel][m * 128 + ci];
            if (wv < P) {
                const _Float16* g = inPad +
                    (((size_t)((n * PD + (d + dz)) * PH + (h + dy))) * P + wv) * 128 + ci;
                async_g2l_b128(g, l);
            } else {
                v8h z = {};
                *(v8h*)l = z;              // ds_store, covered by barrier's dscnt
            }
        }
    };

    stageA(0, 0);
    wait_asynccnt0();
    __syncthreads();

    for (int t = 0; t < 27; ++t) {
        const int cur = t & 1;
        if (t + 1 < 27) stageA(t + 1, cur ^ 1);   // prefetch next tap

        if (wave < nCoT) {
            const _Float16* wb = wfrag + (((size_t)t * nCoT + wave) * 4) * 512;

            // batch-load all fragments, then issue WMMAs back-to-back
            v16h bf[4];
#pragma unroll
            for (int cib = 0; cib < 4; ++cib)
                bf[cib] = *(const v16h*)(wb + (size_t)cib * 512 + lane * 16);

            v16h af[4][MT];
#pragma unroll
            for (int cib = 0; cib < 4; ++cib)
#pragma unroll
                for (int mt = 0; mt < MT; ++mt) {
                    int m  = mt * 16 + (lane & 15);
                    int kb = cib * 32 + ((lane & 16) ? 16 : 0);
                    const _Float16* base = &Alds[cur][m * 128 + kb];
                    v8h lo = *(const v8h*)(base);
                    v8h hi = *(const v8h*)(base + 8);
                    af[cib][mt] = __builtin_shufflevector(lo, hi,
                        0, 1, 2, 3, 4, 5, 6, 7, 8, 9, 10, 11, 12, 13, 14, 15);
                }

#pragma unroll
            for (int cib = 0; cib < 4; ++cib)
#pragma unroll
                for (int mt = 0; mt < MT; ++mt)
                    acc[mt] = __builtin_amdgcn_wmma_f32_16x16x32_f16(
                        false, af[cib][mt], false, bf[cib], (short)0, acc[mt],
                        false, false);

            if (t + 1 < 27) {  // pull next tap's weights toward the WGP
                const _Float16* wn = wfrag + (((size_t)(t + 1) * nCoT + wave) * 4) * 512;
                __builtin_prefetch(wn + (size_t)lane * 64, 0, 0);
            }
        }

        wait_asynccnt0();      // own prefetch of t+1 landed in LDS
        __syncthreads();       // all waves: compute(t) done + buf(t+1) ready
    }

    // ---- epilogue: C/D layout: n = lane%16 (co), m = vgpr + (lane>=16 ? 8:0) ----
    if (wave < nCoT) {
        const int co = wave * 16 + (lane & 15);
#pragma unroll
        for (int mt = 0; mt < MT; ++mt)
#pragma unroll
            for (int v = 0; v < 8; ++v) {
                int w = mt * 16 + v + ((lane & 16) ? 8 : 0);
                if (w >= W) continue;
                float x = acc[mt][v];
                if (relu) x = fmaxf(x, 0.f);
                if (outF32) {
                    if (co < Co)
                        outF32[((((size_t)(n * Co + co) * D + d) * H + h) * W) + w] = x;
                } else {
                    size_t vox = ((size_t)((n * PD + (d + 1)) * PH + (h + 1))) * P + (w + 1);
                    outPad[vox * 128 + co] = (_Float16)x;
                }
            }
    }
}

// ---------------------------------------------------------------------------
// Host-side orchestration
// ---------------------------------------------------------------------------
extern "C" void kernel_launch(void* const* d_in, const int* in_sizes, int n_in,
                              void* d_out, int out_size, void* d_ws, size_t ws_size,
                              hipStream_t stream) {
    (void)in_sizes; (void)n_in; (void)out_size; (void)ws_size;

    const float* feat[3]  = { (const float*)d_in[0], (const float*)d_in[1], (const float*)d_in[2] };
    const float* cls_w[4] = { (const float*)d_in[3],  (const float*)d_in[7],
                              (const float*)d_in[11], (const float*)d_in[15] };
    const float* cls_b[4] = { (const float*)d_in[4],  (const float*)d_in[8],
                              (const float*)d_in[12], (const float*)d_in[16] };
    const float* reg_w[4] = { (const float*)d_in[5],  (const float*)d_in[9],
                              (const float*)d_in[13], (const float*)d_in[17] };
    const float* reg_b[4] = { (const float*)d_in[6],  (const float*)d_in[10],
                              (const float*)d_in[14], (const float*)d_in[18] };
    const float* clsout_w = (const float*)d_in[19];
    const float* clsout_b = (const float*)d_in[20];
    const float* regout_w = (const float*)d_in[21];
    const float* regout_b = (const float*)d_in[22];

    float* out = (float*)d_out;

    // ---- workspace layout (all f16) ----
    const size_t WT = 442368;   // 27*8*4*32*16  tower layer frags
    const size_t WO = 110592;   // 27*2*4*32*16  out-conv frags
    _Float16* ws   = (_Float16*)d_ws;
    _Float16* wt[8];
    for (int i = 0; i < 8; ++i) wt[i] = ws + (size_t)i * WT;
    _Float16* wco = ws + 8 * WT;
    _Float16* wro = wco + WO;
    const size_t PADMAX = (size_t)2 * 34 * 34 * 34 * 128 + 4096; // level-0 + slack
    _Float16* bufA = wro + WO;
    _Float16* bufB = bufA + PADMAX;
    _Float16* bufC = bufB + PADMAX;

    // ---- weight conversion (fragment order) ----
    {
        int blkT = (27 * 8 * 2048 + 255) / 256;
        int blkO = (27 * 2 * 2048 + 255) / 256;
        for (int i = 0; i < 4; ++i) {
            wconv_frag<<<blkT, 256, 0, stream>>>(cls_w[i], wt[i],     128, 8);
            wconv_frag<<<blkT, 256, 0, stream>>>(reg_w[i], wt[4 + i], 128, 8);
        }
        wconv_frag<<<blkO, 256, 0, stream>>>(clsout_w, wco, 15, 2);
        wconv_frag<<<blkO, 256, 0, stream>>>(regout_w, wro, 18, 2);
    }

    const int    dims[3]   = { 32, 16, 8 };
    const size_t clsOff[3] = { 0,       983040,  1105920 };
    const size_t regOff[3] = { 1121280, 2300928, 2448384 };

    auto conv = [&](const _Float16* in, const _Float16* wf, const float* bias,
                    _Float16* outp, float* outf, int Dd, int nCoT, int Co, int relu) {
        int blocks = 2 * Dd * Dd;  // N * D * H
        if (Dd >= 32)
            conv3d_wmma<2><<<blocks, 256, 0, stream>>>(in, wf, bias, outp, outf,
                                                       Dd, Dd, Dd, nCoT, Co, relu);
        else
            conv3d_wmma<1><<<blocks, 256, 0, stream>>>(in, wf, bias, outp, outf,
                                                       Dd, Dd, Dd, nCoT, Co, relu);
    };

    for (int L = 0; L < 3; ++L) {
        const int Dd = dims[L];
        const size_t padN = (size_t)2 * (Dd + 2) * (Dd + 2) * (Dd + 2) * 128;
        const size_t n8   = padN / 8;
        int fb = (int)((n8 + 255) / 256);
        fill0_f16<<<fb, 256, 0, stream>>>(bufA, n8);
        fill0_f16<<<fb, 256, 0, stream>>>(bufB, n8);
        fill0_f16<<<fb, 256, 0, stream>>>(bufC, n8);

        size_t tot = (size_t)2 * 128 * Dd * Dd * Dd;
        im2half<<<(int)((tot + 255) / 256), 256, 0, stream>>>(feat[L], bufA, Dd);

        // classification head
        conv(bufA, wt[0], cls_b[0], bufB, nullptr, Dd, 8, 128, 1);
        conv(bufB, wt[1], cls_b[1], bufC, nullptr, Dd, 8, 128, 1);
        conv(bufC, wt[2], cls_b[2], bufB, nullptr, Dd, 8, 128, 1);
        conv(bufB, wt[3], cls_b[3], bufC, nullptr, Dd, 8, 128, 1);
        conv(bufC, wco, clsout_b, nullptr, out + clsOff[L], Dd, 2, 15, 0);

        // regression head
        conv(bufA, wt[4], reg_b[0], bufB, nullptr, Dd, 8, 128, 1);
        conv(bufB, wt[5], reg_b[1], bufC, nullptr, Dd, 8, 128, 1);
        conv(bufC, wt[6], reg_b[2], bufB, nullptr, Dd, 8, 128, 1);
        conv(bufB, wt[7], reg_b[3], bufC, nullptr, Dd, 8, 128, 1);
        conv(bufC, wro, regout_b, nullptr, out + regOff[L], Dd, 2, 18, 0);
    }
}
